// LatentDynamics_7713761264303
// MI455X (gfx1250) — matrix operand order repaired and checked by reading, compiled, not run on hardware
//
#include <hip/hip_runtime.h>

// ---------------------------------------------------------------------------
// LatentDynamics on MI455X (gfx1250):
//   Phase 1: pack W and z to bf16 in WMMA fragment-native layouts.
//   Phase 2: Zproj = z @ Wz^T, 32768x1024x1024 bf16 WMMA GEMM with
//            double-buffered GLOBAL_LOAD_ASYNC_TO_LDS_B128 staging (ASYNCcnt).
//   Phase 3: sequential recurrence, 4 WGs x 16 batch rows, h in registers
//            (WMMA C/D layout), bf16 h shared via LDS as the A operand,
//            Wh streamed coalesced from L2.
// ---------------------------------------------------------------------------

#define DD 1024
#define TT 512
#define BB 64

typedef __attribute__((ext_vector_type(16))) __bf16 bf16x16;
typedef __attribute__((ext_vector_type(8)))  __bf16 bf16x8;
typedef __attribute__((ext_vector_type(8)))  float  f32x8;
typedef __attribute__((ext_vector_type(4)))  float  f32x4;

static __device__ __forceinline__ bf16x16 make_frag(bf16x8 lo, bf16x8 hi) {
  bf16x16 r;
#pragma unroll
  for (int i = 0; i < 8; ++i) { r[i] = lo[i]; r[i + 8] = hi[i]; }
  return r;
}

static __device__ __forceinline__ f32x8 wmma_bf16(bf16x16 a, bf16x16 b, f32x8 c) {
  return __builtin_amdgcn_wmma_f32_16x16x32_bf16(false, a, false, b,
                                                 (short)0, c, false, false);
}

// Async DMA: 16B per lane, global -> LDS, tracked by ASYNCcnt.
static __device__ __forceinline__ void async_copy16(unsigned lds_off, const void* gptr) {
  asm volatile("global_load_async_to_lds_b128 %0, %1, off"
               :: "v"(lds_off), "v"(gptr) : "memory");
}
static __device__ __forceinline__ void wait_async0() {
  asm volatile("s_wait_asynccnt 0x0" ::: "memory");
}

// ---------------------------------------------------------------------------
// Kernel 0a: pack W into bf16, B-fragment-native:
//   packH[(kb*1024 + c)*32 + k] = bf16(W[c*2048 + kb*32 + k])        (Wh^T)
//   packZ[(kb*1024 + c)*32 + k] = bf16(W[c*2048 + 1024 + kb*32 + k]) (Wz^T)
// B-fragment for (col c, k-block kb) = two contiguous 16B loads.
// ---------------------------------------------------------------------------
__global__ void ld_pack_w(const float* __restrict__ W,
                          __bf16* __restrict__ packH,
                          __bf16* __restrict__ packZ) {
  int tid = blockIdx.x * blockDim.x + threadIdx.x;   // 0..32767
  int kb = tid >> 10;
  int c  = tid & 1023;
  const float* srcH = W + (size_t)c * 2048 + kb * 32;
  const float* srcZ = srcH + 1024;
  __bf16* dH = packH + ((size_t)kb * 1024 + c) * 32;
  __bf16* dZ = packZ + ((size_t)kb * 1024 + c) * 32;
#pragma unroll
  for (int k = 0; k < 32; ++k) {
    dH[k] = (__bf16)srcH[k];
    dZ[k] = (__bf16)srcZ[k];
  }
}

// ---------------------------------------------------------------------------
// Kernel 0b: pack z into bf16, A-fragment-native 1KB blocks:
//   Apack[((rb*32 + kb)*16 + m)*32 + k] = bf16(z[(rb*16 + m)*1024 + kb*32 + k])
// ---------------------------------------------------------------------------
__global__ void ld_pack_z(const float* __restrict__ z, __bf16* __restrict__ Apack) {
  int tid = blockIdx.x * blockDim.x + threadIdx.x;   // 0..1048575
  int m  = tid & 15;
  int kb = (tid >> 4) & 31;
  int rb = tid >> 9;                                 // 0..2047
  const float* src = z + ((size_t)rb * 16 + m) * DD + kb * 32;
  __bf16* dst = Apack + (((size_t)rb * 32 + kb) * 16 + m) * 32;
#pragma unroll
  for (int i = 0; i < 4; ++i) {
    f32x4 a = *(const f32x4*)(src + i * 8);
    f32x4 b = *(const f32x4*)(src + i * 8 + 4);
    bf16x8 o;
#pragma unroll
    for (int q = 0; q < 4; ++q) { o[q] = (__bf16)a[q]; o[q + 4] = (__bf16)b[q]; }
    *(bf16x8*)(dst + i * 8) = o;
  }
}

// ---------------------------------------------------------------------------
// Kernel 1: Zproj[r, n] = sum_k z[r, k] * W[n, 1024 + k]
//   M = 32768, N = 1024, K = 1024. Block tile 128x128, BK = 32.
//   Double-buffered async-DMA staging: stage = [A 8KB][B 8KB], 2 buffers.
//   8 waves, each 32(M) x 64(N) = 2x4 accum tiles, 8 WMMAs / K-chunk.
// ---------------------------------------------------------------------------
__global__ void __launch_bounds__(256)
ld_zproj_gemm(const __bf16* __restrict__ Apack,
              const __bf16* __restrict__ packZ,
              float* __restrict__ zp) {
  __shared__ __bf16 smem[2 * 8192];   // 2 x 16KB stages

  const int tid  = threadIdx.x;
  const int lane = tid & 31, wid = tid >> 5;
  const int m15  = lane & 15, hi = lane >> 4;
  const int wm   = wid & 3;           // 4 M-subtiles of 32 rows
  const int wn   = wid >> 2;          // 2 N-subtiles of 64 cols
  const int Mbase = blockIdx.y * 128;
  const int Nbase = blockIdx.x * 128;

  const unsigned sbase = (unsigned)(uintptr_t)&smem[0];
  const char* aSrc = (const char*)Apack;
  const char* bSrc = (const char*)packZ;
  const int f0 = wid * 2048 + lane * 16;   // flat byte offset in 16KB stage

  auto issue_chunk = [&](int kc, int buf) {
#pragma unroll
    for (int j = 0; j < 4; ++j) {
      int fo = f0 + j * 512;
      const char* g;
      if (wid < 4) {                       // A region: fo in [0, 8KB)
        int rb = fo >> 10, inner = fo & 1023;
        g = aSrc + (((size_t)(Mbase >> 4) + rb) * 32 + kc) * 1024 + inner;
      } else {                             // B region: fo-8KB in [0, 8KB)
        g = bSrc + ((size_t)kc * 1024 + Nbase) * 64 + (fo - 8192);
      }
      async_copy16(sbase + buf * 16384 + fo, g);
    }
  };

  f32x8 acc[2][4] = {};
  issue_chunk(0, 0);

  for (int kc = 0; kc < 32; ++kc) {
    const int buf = kc & 1;
    wait_async0();                 // my copies for this stage are done
    __syncthreads();               // everyone's are; stage is published
    if (kc < 31) issue_chunk(kc + 1, buf ^ 1);   // DMA next stage under WMMA

    const __bf16* Ab = &smem[buf * 8192];
    const __bf16* Bb = Ab + 4096;

    bf16x16 bfr[4];
#pragma unroll
    for (int nt = 0; nt < 4; ++nt) {
      const __bf16* bp = Bb + (wn * 64 + nt * 16 + m15) * 32 + hi * 8;
      bfr[nt] = make_frag(*(const bf16x8*)bp, *(const bf16x8*)(bp + 16));
    }
#pragma unroll
    for (int mt = 0; mt < 2; ++mt) {
      const __bf16* ap = Ab + (wm * 2 + mt) * 512 + m15 * 32 + hi * 8;
      bf16x16 af = make_frag(*(const bf16x8*)ap, *(const bf16x8*)(ap + 16));
#pragma unroll
      for (int nt = 0; nt < 4; ++nt)
        acc[mt][nt] = wmma_bf16(af, bfr[nt], acc[mt][nt]);
    }
  }

#pragma unroll
  for (int mt = 0; mt < 2; ++mt)
#pragma unroll
    for (int nt = 0; nt < 4; ++nt)
#pragma unroll
      for (int v = 0; v < 8; ++v) {
        int row = Mbase + wm * 32 + mt * 16 + v + 8 * hi;   // C/D: M = v + 8*hi
        int col = Nbase + wn * 64 + nt * 16 + m15;
        zp[(size_t)row * DD + col] = acc[mt][nt][v];
      }
}

// ---------------------------------------------------------------------------
// Kernel 2: recurrence. 4 WGs x 16 batch rows, 1024 threads (32 waves).
//   Wave w owns columns [32w, 32w+32): 2 accum tiles, K = 1024 over Wh.
//   h kept in f32 registers (C/D layout); bf16 h shared in LDS (A operand).
// ---------------------------------------------------------------------------
__global__ void __launch_bounds__(1024)
ld_recurrence(const float* __restrict__ h0, const float* __restrict__ z,
              const float* __restrict__ zp, const __bf16* __restrict__ packH,
              const float* __restrict__ bias, const float* __restrict__ alpha,
              float* __restrict__ out) {
  __shared__ __bf16 X[16 * 1032];   // bf16 h, pitch 1032 (16B-aligned rows, bank-safe)

  const int tid  = threadIdx.x;
  const int lane = tid & 31, wid = tid >> 5;
  const int m15  = lane & 15, hi = lane >> 4;
  const int g    = blockIdx.x;                 // batch rows [16g, 16g+16)

  const int c0 = wid * 32 + m15;
  const int c1 = c0 + 16;

  float h[2][8];
  float lk[2] = {1.0f - alpha[c0], 1.0f - alpha[c1]};
  float bs[2] = {bias[c0], bias[c1]};

#pragma unroll
  for (int nt = 0; nt < 2; ++nt) {
    int c = nt ? c1 : c0;
#pragma unroll
    for (int v = 0; v < 8; ++v) {
      int r = v + 8 * hi;
      float hv = h0[(size_t)(g * 16 + r) * DD + c];
      h[nt][v] = hv;
      X[r * 1032 + c] = (__bf16)hv;
    }
  }

  for (int t = 0; t < TT; ++t) {
    __syncthreads();                            // X for this step is ready
    f32x8 acc0 = {}, acc1 = {};
#pragma unroll 4
    for (int kb = 0; kb < 32; ++kb) {
      const __bf16* xr = &X[m15 * 1032 + kb * 32 + 8 * hi];
      bf16x16 af = make_frag(*(const bf16x8*)xr, *(const bf16x8*)(xr + 16));
      const __bf16* b0 = packH + ((size_t)kb * 1024 + c0) * 32 + hi * 8;
      const __bf16* b1 = b0 + 16 * 32;
      if (kb < 31) __builtin_prefetch(b0 + 32 * 1024, 0, 1);  // next k-block of Wh
      bf16x16 bf0 = make_frag(*(const bf16x8*)b0, *(const bf16x8*)(b0 + 16));
      bf16x16 bf1 = make_frag(*(const bf16x8*)b1, *(const bf16x8*)(b1 + 16));
      acc0 = wmma_bf16(af, bf0, acc0);
      acc1 = wmma_bf16(af, bf1, acc1);
    }
    __syncthreads();                            // all A reads of X done

#pragma unroll
    for (int nt = 0; nt < 2; ++nt) {
      int c = nt ? c1 : c0;
#pragma unroll
      for (int v = 0; v < 8; ++v) {
        int r = v + 8 * hi;
        size_t idx = ((size_t)(g * 16 + r) * TT + t) * DD + c;
        float pre  = (nt ? acc1[v] : acc0[v]) + zp[idx] + bs[nt];
        float gate = 1.0f / (1.0f + __expf(-pre));
        float u    = h[nt][v] * lk[nt] + gate * z[idx];
        float e2   = __expf(2.0f * u);
        float hn   = (e2 - 1.0f) / (e2 + 1.0f);   // tanh(u), |u| <= ~7: safe
        h[nt][v]   = hn;
        out[idx]   = hn;
        X[r * 1032 + c] = (__bf16)hn;
      }
    }
  }
}

// ---------------------------------------------------------------------------
extern "C" void kernel_launch(void* const* d_in, const int* in_sizes, int n_in,
                              void* d_out, int out_size, void* d_ws, size_t ws_size,
                              hipStream_t stream) {
  (void)in_sizes; (void)n_in; (void)out_size; (void)ws_size;
  const float* h_t   = (const float*)d_in[0];   // (64, 1024)
  const float* z     = (const float*)d_in[1];   // (64, 512, 1024)
  const float* W     = (const float*)d_in[2];   // (1024, 2048)
  const float* b     = (const float*)d_in[3];   // (1024,)
  const float* alpha = (const float*)d_in[4];   // (1024,)
  float* out = (float*)d_out;                   // (64, 512, 1024)

  // Workspace: [packH 2MB][packZ 2MB][Apack 64MB][Zproj 128MB]
  __bf16* packH = (__bf16*)d_ws;
  __bf16* packZ = packH + (size_t)1048576;
  __bf16* Apack = packZ + (size_t)1048576;
  float*  zproj = (float*)((char*)d_ws + (size_t)68 * 1024 * 1024);

  ld_pack_w<<<128, 256, 0, stream>>>(W, packH, packZ);
  ld_pack_z<<<4096, 256, 0, stream>>>(z, Apack);

  dim3 g1(8, 256);                              // N-blocks x M-blocks
  ld_zproj_gemm<<<g1, 256, 0, stream>>>(Apack, packZ, zproj);

  ld_recurrence<<<4, 1024, 0, stream>>>(h_t, z, zproj, packH, b, alpha, out);
}
